// CustomTemporalAttention_57140244906175
// MI455X (gfx1250) — matrix operand order repaired and checked by reading, not compiled, not run on hardware
//
#include <hip/hip_runtime.h>
#include <math.h>

// ---------------------------------------------------------------------------
// Shapes (fixed by the reference)
// ---------------------------------------------------------------------------
#define BB 8
#define TT 1024
#define EE 1024
#define HH 16
#define DD 64          // head dim
#define KDIM 1024
#define NDIM 1024
#define KC 64          // attention key-chunk staged in LDS per block

typedef __attribute__((ext_vector_type(16))) __bf16 v16bf;
typedef __attribute__((ext_vector_type(8)))  float  v8f;

union FragBF {
    v16bf v;
    __bf16 h[16];
    unsigned short s[16];
    uint4 u[2];
};

union Pack8 {
    __bf16 h[8];
    uint4 u;
};

__device__ __forceinline__ unsigned short f2bits(float f) {
    __bf16 b = (__bf16)f;                       // native v_cvt f32->bf16 (RNE)
    return __builtin_bit_cast(unsigned short, b);
}

__device__ __forceinline__ v8f wmma_bf16(v16bf a, v16bf b, v8f c) {
    return __builtin_amdgcn_wmma_f32_16x16x32_bf16(
        /*neg_a=*/false, a, /*neg_b=*/false, b,
        /*c_mod=*/(short)0, c, /*reuse_a=*/false, /*reuse_b=*/false);
}

// Async global -> LDS copy of 16 bytes (gfx1250 ASYNCcnt path).
// VDST = wave-relative LDS byte offset (low 32 bits of the flat shared ptr,
// per ISA aperture mapping); VADDR = 64-bit global address; no SADDR.
__device__ __forceinline__ void async_copy_b128(void* dst_lds, const void* src_g) {
    unsigned int loff = (unsigned int)(unsigned long long)dst_lds;
    asm volatile("global_load_async_to_lds_b128 %0, %1, off"
                 :: "v"(loff), "v"(src_g) : "memory");
}
__device__ __forceinline__ void wait_async() {
    asm volatile("s_wait_asynccnt 0" ::: "memory");
}

// ---------------------------------------------------------------------------
// 16-lane butterfly reductions with DPP16 (no LDS, no DS-counter waits).
//  xor1 -> quad_perm(1,0,3,2)=0xB1, xor2 -> quad_perm(2,3,0,1)=0x4E,
//  xor4 -> ROW_HALF_MIRROR=0x141,   xor8 -> ROW_MIRROR=0x140.
// ---------------------------------------------------------------------------
template<int CTRL>
__device__ __forceinline__ float dpp_xchg(float x) {
    int i = __builtin_bit_cast(int, x);
    int r = __builtin_amdgcn_update_dpp(i, i, CTRL, 0xF, 0xF, true);
    return __builtin_bit_cast(float, r);
}
__device__ __forceinline__ float red_max16(float x) {
    x = fmaxf(x, dpp_xchg<0xB1>(x));
    x = fmaxf(x, dpp_xchg<0x4E>(x));
    x = fmaxf(x, dpp_xchg<0x141>(x));
    x = fmaxf(x, dpp_xchg<0x140>(x));
    return x;
}
__device__ __forceinline__ float red_sum16(float x) {
    x += dpp_xchg<0xB1>(x);
    x += dpp_xchg<0x4E>(x);
    x += dpp_xchg<0x141>(x);
    x += dpp_xchg<0x140>(x);
    return x;
}

// ---------------------------------------------------------------------------
// Elementwise fp32 -> bf16 (8 elements / thread, b128 in, b128 out)
// ---------------------------------------------------------------------------
__global__ __launch_bounds__(256) void cvt_bf16_kernel(
    const float* __restrict__ in, unsigned short* __restrict__ out, int n8)
{
    int i = blockIdx.x * 256 + threadIdx.x;
    if (i >= n8) return;
    const float4* p = reinterpret_cast<const float4*>(in) + (size_t)i * 2;
    float4 f0 = p[0], f1 = p[1];
    Pack8 r;
    r.h[0] = (__bf16)f0.x; r.h[1] = (__bf16)f0.y;
    r.h[2] = (__bf16)f0.z; r.h[3] = (__bf16)f0.w;
    r.h[4] = (__bf16)f1.x; r.h[5] = (__bf16)f1.y;
    r.h[6] = (__bf16)f1.z; r.h[7] = (__bf16)f1.w;
    reinterpret_cast<uint4*>(out)[i] = r.u;
}

// ---------------------------------------------------------------------------
// Fused temporal-bias table:  fused[rel][h], rel in [0, 2046]
// Exactly replicates reference clamp/floor/ceil/lerp for integer rel.
// ---------------------------------------------------------------------------
__global__ __launch_bounds__(256) void fused_bias_kernel(
    const float* __restrict__ tbl,     // [2T-1, H]
    const float* __restrict__ offset,  // [1]
    float* __restrict__ fused)         // [2T-1, H]
{
    int idx = blockIdx.x * 256 + threadIdx.x;
    if (idx >= (2 * TT - 1) * HH) return;
    int r = idx >> 4, h = idx & 15;
    float bounded = tanhf(offset[0]) * 0.5f;
    float adj = fminf(fmaxf((float)r + bounded, 0.0f), 2.0f * TT - 2.0f);
    float lo = floorf(adj);
    float hi = ceilf(adj);
    float w  = adj - lo;
    fused[idx] = tbl[(int)lo * HH + h] * (1.0f - w) + tbl[(int)hi * HH + h] * w;
}

// ---------------------------------------------------------------------------
// Generic GEMM:  C[M,N] = (A[M,K] @ W[N,K]^T + bias[N]) * scale
//   bf16 operands, f32 accumulate; W-tile (64x64, 8 KB) staged in LDS via
//   async global->LDS loads, double-buffered (all 8 waves share it).
// MODE 0: fp32 out, row-major [M,N]
// MODE 1: bf16 out, [B,H,T,D]            (Q, 1/sqrt(D) folded via scale)
// MODE 2: bf16 out, [B,H,T,D]            (K)
// MODE 3: bf16 out, [B,H,D,T] transposed (V)
// Wave = 16 rows x 64 cols; block = 8 waves stacked in M (128 x 64).
// ---------------------------------------------------------------------------
template<int MODE>
__global__ __launch_bounds__(256) void gemm_bf16_wmma(
    const unsigned short* __restrict__ A,   // [M,K] bf16
    const unsigned short* __restrict__ W,   // [N,K] bf16
    const float* __restrict__ bias,         // [N]  fp32
    void* __restrict__ out,
    float scale)
{
    __shared__ unsigned short ldsW[2][64 * 64];   // [col][k] 8 KB per buffer

    const int tid  = threadIdx.x;
    const int wave = tid >> 5;
    const int lane = tid & 31;
    const int hf   = lane >> 4;   // which 16-lane half
    const int nl   = lane & 15;
    const int row0 = blockIdx.y * 128 + wave * 16;
    const int col0 = blockIdx.x * 64;

    // two async b128 per thread stage the whole 64x64 bf16 W tile
    auto stageW = [&](int k0, int bufi) {
        #pragma unroll
        for (int i = 0; i < 2; ++i) {
            int off = tid * 16 + i * 8;          // 8 bf16 = 16B
            int c = off >> 6, kk = off & 63;
            async_copy_b128(&ldsW[bufi][off],
                            W + (size_t)(col0 + c) * KDIM + k0 + kk);
        }
    };

    v8f acc[4] = {};
    stageW(0, 0);
    wait_async();
    __syncthreads();

    int buf = 0;
    for (int k0 = 0; k0 < KDIM; k0 += 64) {
        if (k0 + 64 < KDIM) stageW(k0 + 64, buf ^ 1);   // overlap DMA w/ compute

        #pragma unroll
        for (int hk = 0; hk < 2; ++hk) {
            const int kk0 = hk * 32;
            // A fragment: lane = row (row0+nl); elem i -> k = hf*16 + i
            FragBF a;
            const unsigned short* arow =
                A + (size_t)(row0 + nl) * KDIM + k0 + kk0;
            a.u[0] = *reinterpret_cast<const uint4*>(arow + hf * 16);
            a.u[1] = *reinterpret_cast<const uint4*>(arow + hf * 16 + 8);

            // B fragments from LDS: lane = col n; elem i -> k = hf*16 + i
            #pragma unroll
            for (int cg = 0; cg < 4; ++cg) {
                FragBF b;
                const unsigned short* p =
                    &ldsW[buf][(cg * 16 + nl) * 64 + kk0 + hf * 16];
                b.u[0] = *reinterpret_cast<const uint4*>(p);
                b.u[1] = *reinterpret_cast<const uint4*>(p + 8);
                acc[cg] = wmma_bf16(a.v, b.v, acc[cg]);
            }
        }

        wait_async();
        __syncthreads();
        buf ^= 1;
    }

    // Epilogue. C fragment: VGPR j -> row (j + hf*8), col = lane&15 (+tile)
    #pragma unroll
    for (int cg = 0; cg < 4; ++cg) {
        #pragma unroll
        for (int j = 0; j < 8; ++j) {
            int r = row0 + j + hf * 8;
            int c = col0 + cg * 16 + nl;
            float v = (acc[cg][j] + bias[c]) * scale;
            if (MODE == 0) {
                reinterpret_cast<float*>(out)[(size_t)r * NDIM + c] = v;
            } else {
                int b = r >> 10, t = r & (TT - 1);
                int h = c >> 6,  d = c & (DD - 1);
                unsigned short bv = f2bits(v);
                if (MODE == 3) // [B,H,D,T]
                    reinterpret_cast<unsigned short*>(out)
                        [(((size_t)(b * HH + h) * DD + d) << 10) + t] = bv;
                else           // [B,H,T,D]
                    reinterpret_cast<unsigned short*>(out)
                        [(((size_t)(b * HH + h) * TT + t) << 6) + d] = bv;
            }
        }
    }
}

// ---------------------------------------------------------------------------
// Flash attention, block-cooperative:
//   block = 8 waves = one (b, h, 128-query slab); wave = 16 queries.
//   K/V chunks (KC=64 keys) staged in LDS via async global->LDS loads,
//   double-buffered so the DMA of chunk i+1 overlaps WMMA on chunk i.
// ---------------------------------------------------------------------------
__global__ __launch_bounds__(256) void attn_kernel(
    const unsigned short* __restrict__ qbf,   // [B,H,T,D] bf16 (pre-scaled)
    const unsigned short* __restrict__ kbf,   // [B,H,T,D] bf16
    const unsigned short* __restrict__ vtbf,  // [B,H,D,T] bf16
    const float* __restrict__ fused,          // [2T-1, H]
    unsigned short* __restrict__ attnout)     // [B*T, E] bf16
{
    __shared__ unsigned short ldsK[2][KC * DD];   // [key][d]   8 KB each
    __shared__ unsigned short ldsV[2][DD * KC];   // [d][key]   8 KB each
    __shared__ unsigned short plds[8][16 * KC];   // per-wave P (16x64) 2 KB each

    const int tid  = threadIdx.x;
    const int wave = tid >> 5;
    const int lane = tid & 31;
    const int hf   = lane >> 4;
    const int nl   = lane & 15;

    const int blk = blockIdx.x;          // B*H*(T/128) = 1024
    const int b   = blk >> 7;            // H*(T/128) = 128 per batch
    const int rem = blk & 127;
    const int h   = rem >> 3;            // T/128 = 8 slabs
    const int q0  = ((rem & 7) << 7) + wave * 16;

    const unsigned short* qh = qbf  + (size_t)(b * HH + h) * TT * DD;
    const unsigned short* kh = kbf  + (size_t)(b * HH + h) * TT * DD;
    const unsigned short* vh = vtbf + (size_t)(b * HH + h) * DD * TT;

    // stage one KC-chunk of K ([key][d], contiguous) and V ([d][key] rows)
    auto stage = [&](int c0, int bufi) {
        #pragma unroll
        for (int i = 0; i < 2; ++i) {
            int off = tid * 16 + i * 8;              // element offset, 8 elems=16B
            async_copy_b128(&ldsK[bufi][off], kh + (size_t)c0 * DD + off);
            int d = off >> 6, kl = off & (KC - 1);   // KC == 64
            async_copy_b128(&ldsV[bufi][off], vh + (size_t)d * TT + c0 + kl);
        }
    };

    // Q A-fragments (16x64 = two 16x32): elem i -> d = kk*32+hf*16+i
    FragBF aq[2];
    #pragma unroll
    for (int kk = 0; kk < 2; ++kk)
        #pragma unroll
        for (int g = 0; g < 2; ++g)
            aq[kk].u[g] = *reinterpret_cast<const uint4*>(
                qh + (size_t)(q0 + nl) * DD + kk * 32 + g * 8 + hf * 16);

    float mrow[8], lrow[8];
    v8f o[4] = {};
    #pragma unroll
    for (int j = 0; j < 8; ++j) { mrow[j] = -1e30f; lrow[j] = 0.0f; }

    stage(0, 0);
    wait_async();
    __syncthreads();

    int buf = 0;
    for (int c0 = 0; c0 < TT; c0 += KC) {
        if (c0 + KC < TT) stage(c0 + KC, buf ^ 1);   // overlap DMA with compute

        // ---- scores: four 16x16 tiles over this 64-key chunk ----
        v8f s[4];
        #pragma unroll
        for (int st = 0; st < 4; ++st) {
            v8f a = {};
            #pragma unroll
            for (int kk = 0; kk < 2; ++kk) {
                FragBF bk; // B[k=d,n=key]: lane nl = key st*16+nl, elem i -> d=kk*32+hf*16+i
                const unsigned short* p =
                    &ldsK[buf][(st * 16 + nl) * DD + kk * 32 + hf * 16];
                bk.u[0] = *reinterpret_cast<const uint4*>(p);
                bk.u[1] = *reinterpret_cast<const uint4*>(p + 8);
                a = wmma_bf16(aq[kk].v, bk.v, a);
            }
            s[st] = a;
        }
        // ---- temporal bias: one base per chunk, compile-time IOFFSETs ----
        // rel(st,j) = (q0+j+hf*8) - (c0+st*16+nl) + (TT-1)
        //           = base_rel + (3-st)*16 + j, base_rel >= 0 always.
        {
            const float* bb = fused
                + (size_t)((q0 + hf * 8) - (c0 + 48 + nl) + (TT - 1)) * HH + h;
            #pragma unroll
            for (int st = 0; st < 4; ++st)
                #pragma unroll
                for (int j = 0; j < 8; ++j)
                    s[st][j] += bb[((3 - st) * 16 + j) * HH];
        }
        // ---- online softmax (DPP butterfly reductions) ----
        float mc[8];
        #pragma unroll
        for (int j = 0; j < 8; ++j)
            mc[j] = red_max16(
                fmaxf(fmaxf(s[0][j], s[1][j]), fmaxf(s[2][j], s[3][j])));
        float alpha[8];
        #pragma unroll
        for (int j = 0; j < 8; ++j) {
            float mn = fmaxf(mrow[j], mc[j]);
            alpha[j] = __expf(mrow[j] - mn);
            mrow[j] = mn;
        }
        float lc[8];
        #pragma unroll
        for (int j = 0; j < 8; ++j) {
            float p0 = __expf(s[0][j] - mrow[j]);
            float p1 = __expf(s[1][j] - mrow[j]);
            float p2 = __expf(s[2][j] - mrow[j]);
            float p3 = __expf(s[3][j] - mrow[j]);
            s[0][j] = p0; s[1][j] = p1; s[2][j] = p2; s[3][j] = p3;
            lc[j] = red_sum16((p0 + p1) + (p2 + p3));
        }
        #pragma unroll
        for (int j = 0; j < 8; ++j) lrow[j] = lrow[j] * alpha[j] + lc[j];
        #pragma unroll
        for (int t = 0; t < 4; ++t)
            #pragma unroll
            for (int j = 0; j < 8; ++j) o[t][j] *= alpha[j];

        // ---- re-layout P (16x64): C-frags -> LDS -> two A-frags ----
        unsigned short* pw = plds[wave];
        __builtin_amdgcn_wave_barrier();
        #pragma unroll
        for (int st = 0; st < 4; ++st)
            #pragma unroll
            for (int j = 0; j < 8; ++j)
                pw[(j + hf * 8) * KC + st * 16 + nl] = f2bits(s[st][j]);
        __builtin_amdgcn_wave_barrier();
        FragBF ap[2];  // elem i -> k = sub*32 + hf*16 + i, row = nl
        #pragma unroll
        for (int sub = 0; sub < 2; ++sub)
            #pragma unroll
            for (int g = 0; g < 2; ++g)
                ap[sub].u[g] = *reinterpret_cast<const uint4*>(
                    pw + nl * KC + sub * 32 + g * 8 + hf * 16);
        __builtin_amdgcn_wave_barrier();

        // ---- O += P @ V   (4 d-tiles x 2 key-subchunks) ----
        #pragma unroll
        for (int t = 0; t < 4; ++t) {
            #pragma unroll
            for (int sub = 0; sub < 2; ++sub) {
                FragBF bv; // B[k=key,n=d]: lane nl = d t*16+nl, elem i -> key sub*32+hf*16+i
                const unsigned short* p =
                    &ldsV[buf][(t * 16 + nl) * KC + sub * 32 + hf * 16];
                bv.u[0] = *reinterpret_cast<const uint4*>(p);
                bv.u[1] = *reinterpret_cast<const uint4*>(p + 8);
                o[t] = wmma_bf16(ap[sub].v, bv.v, o[t]);
            }
        }

        wait_async();        // next chunk's DMA landed
        __syncthreads();     // everyone done reading buf before it is reused
        buf ^= 1;
    }

    // ---- epilogue: normalize, store bf16 [B*T, E] (feeds final GEMM) ----
    #pragma unroll
    for (int t = 0; t < 4; ++t)
        #pragma unroll
        for (int j = 0; j < 8; ++j) {
            int row = q0 + j + hf * 8;
            int col = h * DD + t * 16 + nl;
            attnout[((size_t)(b * TT + row)) * EE + col] =
                f2bits(o[t][j] / lrow[j]);
        }
}

// ---------------------------------------------------------------------------
// Launch
// ---------------------------------------------------------------------------
extern "C" void kernel_launch(void* const* d_in, const int* in_sizes, int n_in,
                              void* d_out, int out_size, void* d_ws, size_t ws_size,
                              hipStream_t stream) {
    (void)in_sizes; (void)n_in; (void)out_size; (void)ws_size;

    const float* query = (const float*)d_in[0];
    const float* key_  = (const float*)d_in[1];
    const float* value = (const float*)d_in[2];
    const float* Wq = (const float*)d_in[3];
    const float* bq = (const float*)d_in[4];
    const float* Wk = (const float*)d_in[5];
    const float* bk = (const float*)d_in[6];
    const float* Wv = (const float*)d_in[7];
    const float* bv = (const float*)d_in[8];
    const float* Wo = (const float*)d_in[9];
    const float* bo = (const float*)d_in[10];
    const float* bias_table = (const float*)d_in[11];
    const float* offset     = (const float*)d_in[12];

    char* ws = (char*)d_ws;
    const size_t MB = 1u << 20;
    unsigned short* qbf   = (unsigned short*)(ws);              // 16 MB [B,H,T,D]
    unsigned short* kbf   = (unsigned short*)(ws + 16 * MB);    // 16 MB [B,H,T,D]
    unsigned short* vtbf  = (unsigned short*)(ws + 32 * MB);    // 16 MB [B,H,D,T]
    unsigned short* attnb = (unsigned short*)(ws + 48 * MB);    // 16 MB [B*T,E]
    float*          fused = (float*)         (ws + 64 * MB);    // 128 KB pad
    unsigned short* xqbf  = (unsigned short*)(ws + 65 * MB);    // 16 MB
    unsigned short* xkbf  = (unsigned short*)(ws + 81 * MB);    // 16 MB
    unsigned short* xvbf  = (unsigned short*)(ws + 97 * MB);    // 16 MB
    unsigned short* wqbf  = (unsigned short*)(ws + 113 * MB);   // 2 MB
    unsigned short* wkbf  = (unsigned short*)(ws + 115 * MB);   // 2 MB
    unsigned short* wvbf  = (unsigned short*)(ws + 117 * MB);   // 2 MB
    unsigned short* wobf  = (unsigned short*)(ws + 119 * MB);   // 2 MB

    // 0) one-shot bf16 conversions (inputs + weights)
    const int N8_IN = (BB * TT * EE) / 8;   // 1,048,576
    const int N8_W  = (EE * EE) / 8;        // 131,072
    cvt_bf16_kernel<<<N8_IN / 256, 256, 0, stream>>>(query, xqbf, N8_IN);
    cvt_bf16_kernel<<<N8_IN / 256, 256, 0, stream>>>(key_,  xkbf, N8_IN);
    cvt_bf16_kernel<<<N8_IN / 256, 256, 0, stream>>>(value, xvbf, N8_IN);
    cvt_bf16_kernel<<<N8_W  / 256, 256, 0, stream>>>(Wq, wqbf, N8_W);
    cvt_bf16_kernel<<<N8_W  / 256, 256, 0, stream>>>(Wk, wkbf, N8_W);
    cvt_bf16_kernel<<<N8_W  / 256, 256, 0, stream>>>(Wv, wvbf, N8_W);
    cvt_bf16_kernel<<<N8_W  / 256, 256, 0, stream>>>(Wo, wobf, N8_W);

    // 1) fused temporal-bias table
    fused_bias_kernel<<<((2 * TT - 1) * HH + 255) / 256, 256, 0, stream>>>(
        bias_table, offset, fused);

    // 2) projections (scale = 1/sqrt(D) folded into Q)
    dim3 ggrid(NDIM / 64, (BB * TT) / 128);
    gemm_bf16_wmma<1><<<ggrid, 256, 0, stream>>>(xqbf, wqbf, bq, qbf,  0.125f);
    gemm_bf16_wmma<2><<<ggrid, 256, 0, stream>>>(xkbf, wkbf, bk, kbf,  1.0f);
    gemm_bf16_wmma<3><<<ggrid, 256, 0, stream>>>(xvbf, wvbf, bv, vtbf, 1.0f);

    // 3) flash attention: one block per (b, h, 128-query slab)
    attn_kernel<<<BB * HH * (TT / 128), 256, 0, stream>>>(
        qbf, kbf, vtbf, fused, attnb);

    // 4) output projection -> fp32 d_out
    gemm_bf16_wmma<0><<<ggrid, 256, 0, stream>>>(attnb, wobf, bo, (float*)d_out, 1.0f);
}
